// Transformer_1726576854635
// MI455X (gfx1250) — matrix operand order repaired and checked by reading, not compiled, not run on hardware
//
#include <hip/hip_runtime.h>
#include <math.h>

// ---------------------------------------------------------------------------
// Llama-style transformer forward on gfx1250 (CDNA5, wave32, WMMA).
// GEMMs: bf16 WMMA 16x16x32 with f32 accumulate. A and B tiles staged through
// LDS as packed-bf16 pairs (coalesced global_load_b128 fills, in-flight
// f32->bf16 cvt_pk, ds_load_2addr_b64 fragment gathers). K-step = 64: one
// LDS fill feeds 8 WMMAs per wave. Uniform full-tile fast path avoids exec
// juggling in all the big GEMMs (only the M=2 logits GEMM takes guards).
// Attention: f32 flash-style online softmax. Residuals fused in epilogue.
// ---------------------------------------------------------------------------

typedef __attribute__((ext_vector_type(16))) __bf16 v16bf;
typedef __attribute__((ext_vector_type(8)))  float  v8f;

#define DIM_    1024
#define NHEADS_ 16
#define HDIM_   64
#define NLAYER_ 4
#define VOCAB_  32000
#define HID_    2816
#define BSZ_    2
#define SEQ_    1024
#define TOK_    (BSZ_*SEQ_)
#define EPS_    1e-6f
#define THETA_  10000.0f

// Pack two f32 into one dword of bf16 pairs (lowers to v_cvt_pk_bf16_f32).
__device__ inline unsigned pack_bf16(float x, float y)
{
    union { __bf16 h[2]; unsigned u; } p;
    p.h[0] = (__bf16)x;
    p.h[1] = (__bf16)y;
    return p.u;
}

// Fragment viewed either as 8 packed dwords or as the WMMA bf16 vector.
union FragBF { unsigned u[8]; v16bf v; };

// ---------------------------------------------------------------------------
// Generic GEMM: C[M,N] = A[M,K] @ B[K,N] (+ Res), all fp32 row-major.
// Block = 128 threads = 4 waves; wave computes a 16(M)x64(N) tile.
// Requires K%64==0, N%64==0; M guarded only on the (uniform) partial path.
// LDS staging (packed bf16 pairs, dword = {k, k+1}), K-step 64:
//   As[m][kp]   m=0..63, kp=0..31, stride 34 dwords
//   Bs[n][kp]   n=0..63, kp=0..31, stride 34 dwords   (B transposed on fill)
// Fragment layouts, contiguous k order in A and B (slice s = 0,1 -> kp base 16s):
//   A 16x32 bf16 : lane half g=lane>>4, m=lane&15; vgpr r holds kp =
//                  16s + 8g + r.
//   B 32x16 bf16 : n = lane&15; vgpr r holds kp = 16s + 8g + r.
//   C 16x16 f32  : vgpr r holds M = r + 8g, N = lane&15.
// ---------------------------------------------------------------------------
__global__ __launch_bounds__(128)
void gemm_wmma_bf16(const float* __restrict__ A, const float* __restrict__ Bm,
                    const float* Res, float* C, int M, int N, int K)
{
    __shared__ unsigned As[64 * 34];
    __shared__ unsigned Bs[64 * 34];

    const int tid   = (int)threadIdx.x;
    const int lane  = tid & 31;
    const int wave  = tid >> 5;
    const int group = lane >> 4;     // 0: lanes 0-15, 1: lanes 16-31
    const int mi    = lane & 15;
    const int mBase = blockIdx.x * 64;
    const int nBase = blockIdx.y * 64;
    const bool fullTile = (mBase + 64 <= M);   // uniform across the block

    v8f zero = {};
    v8f acc[4] = { zero, zero, zero, zero };

    for (int k0 = 0; k0 < K; k0 += 64) {
        __syncthreads();

        // ---- Fill A tile: 64 rows x 64 k, coalesced float4 along K ----
        if (fullTile) {
            #pragma unroll
            for (int i = 0; i < 8; ++i) {
                int q = tid + i * 128;      // 0..1023 : 64 rows * 16 float4
                int r = q >> 4;             // row in tile
                int c = q & 15;             // float4 group along K
                float4 f = *(const float4*)(A + (size_t)(mBase + r) * K + (k0 + 4 * c));
                As[r * 34 + 2 * c]     = pack_bf16(f.x, f.y);
                As[r * 34 + 2 * c + 1] = pack_bf16(f.z, f.w);
            }
        } else {
            #pragma unroll
            for (int i = 0; i < 8; ++i) {
                int q = tid + i * 128;
                int r = q >> 4;
                int c = q & 15;
                int m = mBase + r;
                float4 f = (m < M)
                    ? *(const float4*)(A + (size_t)m * K + (k0 + 4 * c))
                    : make_float4(0.f, 0.f, 0.f, 0.f);
                As[r * 34 + 2 * c]     = pack_bf16(f.x, f.y);
                As[r * 34 + 2 * c + 1] = pack_bf16(f.z, f.w);
            }
        }

        // ---- Fill B tile transposed: 64 k x 64 n, coalesced float4 along N ----
        #pragma unroll
        for (int i = 0; i < 4; ++i) {
            int q  = tid + i * 128;         // 0..511 : 32 kp * 16 n-groups
            int kp = q >> 4;                // k-pair 0..31
            int c  = q & 15;                // n-group of 4
            const float* b0 = Bm + (size_t)(k0 + 2 * kp) * N + (nBase + 4 * c);
            float4 f0 = *(const float4*)b0;        // row k = k0 + 2kp
            float4 f1 = *(const float4*)(b0 + N);  // row k + 1
            int nb = 4 * c;
            Bs[(nb + 0) * 34 + kp] = pack_bf16(f0.x, f1.x);
            Bs[(nb + 1) * 34 + kp] = pack_bf16(f0.y, f1.y);
            Bs[(nb + 2) * 34 + kp] = pack_bf16(f0.z, f1.z);
            Bs[(nb + 3) * 34 + kp] = pack_bf16(f0.w, f1.w);
        }
        __syncthreads();

        // ---- Two 16x16x32 K-slices per fill: 8 WMMAs ----
        const int lr = wave * 16 + mi;
        #pragma unroll
        for (int s = 0; s < 2; ++s) {
            FragBF a;
            #pragma unroll
            for (int r = 0; r < 8; ++r) {
                int kp = 16 * s + 8 * group + r;
                a.u[r] = As[lr * 34 + kp];
            }
            #pragma unroll
            for (int j = 0; j < 4; ++j) {
                FragBF b;
                int nl = 16 * j + mi;
                #pragma unroll
                for (int r = 0; r < 8; ++r)
                    b.u[r] = Bs[nl * 34 + 16 * s + 8 * group + r];
                acc[j] = __builtin_amdgcn_wmma_f32_16x16x32_bf16(
                    false, a.v, false, b.v, (short)0, acc[j], false, false);
            }
        }
    }

    // ---- Epilogue: optional residual add; guarded only on partial tiles ----
    if (fullTile) {
        #pragma unroll
        for (int j = 0; j < 4; ++j) {
            int n = nBase + 16 * j + mi;
            #pragma unroll
            for (int r = 0; r < 8; ++r) {
                size_t idx = (size_t)(mBase + wave * 16 + r + 8 * group) * N + n;
                float v = acc[j][r];
                if (Res) v += Res[idx];
                C[idx] = v;
            }
        }
    } else {
        #pragma unroll
        for (int j = 0; j < 4; ++j) {
            int n = nBase + 16 * j + mi;
            #pragma unroll
            for (int r = 0; r < 8; ++r) {
                int m = mBase + wave * 16 + r + 8 * group;
                if (m < M) {
                    size_t idx = (size_t)m * N + n;
                    float v = acc[j][r];
                    if (Res) v += Res[idx];
                    C[idx] = v;
                }
            }
        }
    }
}

// ---------------------------------------------------------------------------
// Embedding gather: h[t,:] = tok_emb[tokens[t],:]  (float4 copies)
// ---------------------------------------------------------------------------
__global__ void embed_kernel(const int* __restrict__ tokens,
                             const float* __restrict__ emb,
                             float* __restrict__ h)
{
    int t = blockIdx.x;
    int tok = tokens[t];
    const float4* src = (const float4*)(emb + (size_t)tok * DIM_);
    float4* dst = (float4*)(h + (size_t)t * DIM_);
    for (int i = threadIdx.x; i < DIM_ / 4; i += blockDim.x) dst[i] = src[i];
}

// ---------------------------------------------------------------------------
// RMSNorm per row: y = x * rsqrt(mean(x^2)+eps) * w
// ---------------------------------------------------------------------------
__global__ void rmsnorm_kernel(const float* __restrict__ x,
                               const float* __restrict__ w,
                               float* __restrict__ y, int D)
{
    __shared__ float red[32];
    int row = blockIdx.x;
    const float* xr = x + (size_t)row * D;
    float* yr = y + (size_t)row * D;

    float s = 0.f;
    for (int i = threadIdx.x; i < D; i += blockDim.x) { float v = xr[i]; s += v * v; }
    #pragma unroll
    for (int off = 16; off > 0; off >>= 1) s += __shfl_down(s, off, 32);
    int lane = threadIdx.x & 31, wv = threadIdx.x >> 5;
    if (lane == 0) red[wv] = s;
    __syncthreads();
    if (wv == 0) {
        s = (lane < ((int)blockDim.x >> 5)) ? red[lane] : 0.f;
        #pragma unroll
        for (int off = 16; off > 0; off >>= 1) s += __shfl_down(s, off, 32);
        if (lane == 0) red[0] = s;
    }
    __syncthreads();
    float inv = rsqrtf(red[0] / (float)D + EPS_);
    for (int i = threadIdx.x; i < D; i += blockDim.x) yr[i] = xr[i] * inv * w[i];
}

// ---------------------------------------------------------------------------
// RoPE (start_pos = 0): rotate pairs of head dims by pos * theta^(-2p/Dh).
// x layout: [T, H*Dh] == [B,S,H,Dh] flattened.
// ---------------------------------------------------------------------------
__global__ void rope_kernel(float* __restrict__ x, int total)
{
    int idx = blockIdx.x * blockDim.x + threadIdx.x;
    if (idx >= total) return;
    const int HP = HDIM_ / 2;
    int p = idx % HP;
    int h = (idx / HP) % NHEADS_;
    int t = idx / (HP * NHEADS_);
    int pos = t % SEQ_;
    float freq = expf(-((2.0f * (float)p) / (float)HDIM_) * logf(THETA_));
    float ang = (float)pos * freq;
    float c = cosf(ang), s = sinf(ang);
    float* ptr = x + (size_t)t * DIM_ + h * HDIM_ + 2 * p;
    float x0 = ptr[0], x1 = ptr[1];
    ptr[0] = x0 * c - x1 * s;
    ptr[1] = x0 * s + x1 * c;
}

// ---------------------------------------------------------------------------
// Causal attention, flash-style online softmax in f32.
// One thread per (b, head, query). q/k/v/o layout: [T, H*Dh].
// ---------------------------------------------------------------------------
__global__ void attn_kernel(const float* __restrict__ q,
                            const float* __restrict__ k,
                            const float* __restrict__ v,
                            float* __restrict__ o)
{
    int idx = blockIdx.x * blockDim.x + threadIdx.x;
    if (idx >= BSZ_ * NHEADS_ * SEQ_) return;
    int qi = idx % SEQ_;
    int hh = (idx / SEQ_) % NHEADS_;
    int b  = idx / (SEQ_ * NHEADS_);

    const size_t headOff = (size_t)b * SEQ_ * DIM_ + (size_t)hh * HDIM_;
    const float* qp = q + headOff + (size_t)qi * DIM_;

    float qreg[HDIM_];
    #pragma unroll
    for (int d = 0; d < HDIM_; d += 4) {
        float4 f = *(const float4*)(qp + d);
        qreg[d] = f.x; qreg[d+1] = f.y; qreg[d+2] = f.z; qreg[d+3] = f.w;
    }
    float acc[HDIM_];
    #pragma unroll
    for (int d = 0; d < HDIM_; ++d) acc[d] = 0.f;

    const float scale = 0.125f;               // 1/sqrt(64)
    float m = -INFINITY, l = 0.f;

    for (int j = 0; j <= qi; ++j) {
        const float* kp = k + headOff + (size_t)j * DIM_;
        float s = 0.f;
        #pragma unroll
        for (int d = 0; d < HDIM_; d += 4) {
            float4 f = *(const float4*)(kp + d);
            s += qreg[d]*f.x + qreg[d+1]*f.y + qreg[d+2]*f.z + qreg[d+3]*f.w;
        }
        s *= scale;
        float mnew = fmaxf(m, s);
        float corr = __expf(m - mnew);        // first iter: exp(-inf) = 0
        float p    = __expf(s - mnew);
        l = l * corr + p;
        const float* vp = v + headOff + (size_t)j * DIM_;
        #pragma unroll
        for (int d = 0; d < HDIM_; d += 4) {
            float4 f = *(const float4*)(vp + d);
            acc[d]   = acc[d]  * corr + p * f.x;
            acc[d+1] = acc[d+1]* corr + p * f.y;
            acc[d+2] = acc[d+2]* corr + p * f.z;
            acc[d+3] = acc[d+3]* corr + p * f.w;
        }
        m = mnew;
    }
    float invl = 1.f / l;
    float* op = o + headOff + (size_t)qi * DIM_;
    #pragma unroll
    for (int d = 0; d < HDIM_; ++d) op[d] = acc[d] * invl;
}

// ---------------------------------------------------------------------------
// u = silu(u) * g  (elementwise)
// ---------------------------------------------------------------------------
__global__ void silu_mul_kernel(float* __restrict__ u, const float* __restrict__ g,
                                size_t n)
{
    size_t i = (size_t)blockIdx.x * blockDim.x + threadIdx.x;
    if (i < n) {
        float x = u[i];
        float s = x / (1.f + __expf(-x));
        u[i] = s * g[i];
    }
}

// ---------------------------------------------------------------------------
// Copy the last-position row of each batch into a compact [B, D] buffer.
// ---------------------------------------------------------------------------
__global__ void gather_last_kernel(const float* __restrict__ xn,
                                   float* __restrict__ out2)
{
    int b = blockIdx.x;
    const float4* src = (const float4*)(xn + ((size_t)b * SEQ_ + (SEQ_ - 1)) * DIM_);
    float4* dst = (float4*)(out2 + (size_t)b * DIM_);
    for (int i = threadIdx.x; i < DIM_ / 4; i += blockDim.x) dst[i] = src[i];
}

// ---------------------------------------------------------------------------
// Host-side orchestration (graph-capture safe: only launches on `stream`).
// ---------------------------------------------------------------------------
extern "C" void kernel_launch(void* const* d_in, const int* in_sizes, int n_in,
                              void* d_out, int out_size, void* d_ws, size_t ws_size,
                              hipStream_t stream)
{
    const int*   tokens       = (const int*)  d_in[0];
    // d_in[1] = start_pos (== 0 per harness)
    const float* tok_emb      = (const float*)d_in[2];
    const float* wq           = (const float*)d_in[3];
    const float* wk           = (const float*)d_in[4];
    const float* wv           = (const float*)d_in[5];
    const float* wo           = (const float*)d_in[6];
    const float* w1           = (const float*)d_in[7];
    const float* w2           = (const float*)d_in[8];
    const float* w3           = (const float*)d_in[9];
    const float* attn_norm_w  = (const float*)d_in[10];
    const float* ffn_norm_w   = (const float*)d_in[11];
    const float* final_norm_w = (const float*)d_in[12];
    const float* out_w        = (const float*)d_in[13];
    float* out = (float*)d_out;

    (void)in_sizes; (void)n_in; (void)out_size; (void)ws_size;

    // Workspace carve-up (~96.5 MB of f32).
    float* ws = (float*)d_ws;
    size_t off = 0;
    float* h    = ws + off; off += (size_t)TOK_ * DIM_;
    float* xn   = ws + off; off += (size_t)TOK_ * DIM_;
    float* qb   = ws + off; off += (size_t)TOK_ * DIM_;
    float* kb   = ws + off; off += (size_t)TOK_ * DIM_;
    float* vb   = ws + off; off += (size_t)TOK_ * DIM_;
    float* ob   = ws + off; off += (size_t)TOK_ * DIM_;
    float* ub   = ws + off; off += (size_t)TOK_ * HID_;
    float* gb   = ws + off; off += (size_t)TOK_ * HID_;
    float* last2= ws + off; off += (size_t)BSZ_ * DIM_;

    embed_kernel<<<TOK_, 256, 0, stream>>>(tokens, tok_emb, h);

    const dim3 gDD((TOK_ + 63) / 64, DIM_ / 64);   // [T,1024] outputs
    const dim3 gDH((TOK_ + 63) / 64, HID_ / 64);   // [T,2816] outputs
    const int  ropePairs = TOK_ * NHEADS_ * (HDIM_ / 2);
    const int  attnThreads = BSZ_ * NHEADS_ * SEQ_;

    for (int l = 0; l < NLAYER_; ++l) {
        const float* lwq = wq + (size_t)l * DIM_ * DIM_;
        const float* lwk = wk + (size_t)l * DIM_ * DIM_;
        const float* lwv = wv + (size_t)l * DIM_ * DIM_;
        const float* lwo = wo + (size_t)l * DIM_ * DIM_;
        const float* lw1 = w1 + (size_t)l * DIM_ * HID_;
        const float* lw2 = w2 + (size_t)l * HID_ * DIM_;
        const float* lw3 = w3 + (size_t)l * DIM_ * HID_;

        // Attention block
        rmsnorm_kernel<<<TOK_, 256, 0, stream>>>(h, attn_norm_w + (size_t)l * DIM_, xn, DIM_);
        gemm_wmma_bf16<<<gDD, 128, 0, stream>>>(xn, lwq, nullptr, qb, TOK_, DIM_, DIM_);
        gemm_wmma_bf16<<<gDD, 128, 0, stream>>>(xn, lwk, nullptr, kb, TOK_, DIM_, DIM_);
        gemm_wmma_bf16<<<gDD, 128, 0, stream>>>(xn, lwv, nullptr, vb, TOK_, DIM_, DIM_);
        rope_kernel<<<(ropePairs + 255) / 256, 256, 0, stream>>>(qb, ropePairs);
        rope_kernel<<<(ropePairs + 255) / 256, 256, 0, stream>>>(kb, ropePairs);
        attn_kernel<<<(attnThreads + 127) / 128, 128, 0, stream>>>(qb, kb, vb, ob);
        gemm_wmma_bf16<<<gDD, 128, 0, stream>>>(ob, lwo, h, h, TOK_, DIM_, DIM_);

        // FFN block
        rmsnorm_kernel<<<TOK_, 256, 0, stream>>>(h, ffn_norm_w + (size_t)l * DIM_, xn, DIM_);
        gemm_wmma_bf16<<<gDH, 128, 0, stream>>>(xn, lw1, nullptr, ub, TOK_, HID_, DIM_);
        gemm_wmma_bf16<<<gDH, 128, 0, stream>>>(xn, lw3, nullptr, gb, TOK_, HID_, DIM_);
        size_t nH = (size_t)TOK_ * HID_;
        silu_mul_kernel<<<(unsigned)((nH + 255) / 256), 256, 0, stream>>>(ub, gb, nH);
        gemm_wmma_bf16<<<gDD, 128, 0, stream>>>(ub, lw2, h, h, TOK_, DIM_, HID_);
    }

    // Final norm + last-position logits
    rmsnorm_kernel<<<TOK_, 256, 0, stream>>>(h, final_norm_w, xn, DIM_);
    gather_last_kernel<<<BSZ_, 256, 0, stream>>>(xn, last2);
    const dim3 gOut(1, VOCAB_ / 64);               // M=2 guarded inside
    gemm_wmma_bf16<<<gOut, 128, 0, stream>>>(last2, out_w, nullptr, out, BSZ_, VOCAB_, DIM_);
}
